// SelectMotifAttachment_1623497637905
// MI455X (gfx1250) — matrix operand order repaired and checked by reading, not compiled, not run on hardware
//
#include <hip/hip_runtime.h>
#include <hip/hip_bf16.h>
#include <math.h>

typedef __attribute__((ext_vector_type(8)))  __bf16 v8bf;
typedef __attribute__((ext_vector_type(16))) __bf16 v16bf;
typedef __attribute__((ext_vector_type(8)))  float  v8f;

#define A_ATOMS 10
#define G_MOL   8
#define NODES   (A_ATOMS*G_MOL)    // 80
#define EDGES   (2*A_ATOMS*G_MOL)  // 160
#define HN      64
#define HE      64
#define FN      16
#define FE      8
#define MR      128
#define L1      256
#define L2      128
#define L3      64
#define K1PAD   160   // w_e: 136 -> 160
#define K2PAD   96    // w_n: 80 -> 96
#define THREADS 256

// ---------------- LDS layout (bytes) ----------------
#define OFF_NHB   0                 // bf16 [80][64]  = 10240 (persistent)
#define OFF_NHF   10240             // f32  [80][64]  = 20480 (persistent)
#define OVL       30720             // overlay base
// --- MP-phase overlay ---
#define OFF_WET   (OVL + 0)         // bf16 [64][160] = 20480
#define OFF_WNT   (OVL + 20480)     // bf16 [64][96]  = 12288
#define OFF_BE    (OVL + 32768)     // f32  [64]
#define OFF_BN    (OVL + 33024)     // f32  [64]
#define OFF_NF    (OVL + 33280)     // bf16 [80][16]  = 2560
#define OFF_EF    (OVL + 35840)     // bf16 [160][8]  = 2560
#define OFF_SRC   (OVL + 38400)     // int  [160]
#define OFF_DST   (OVL + 39040)     // int  [160]
#define OFF_CNT   (OVL + 39680)     // int  [80]
#define OFF_INL   (OVL + 40000)     // int  [80][4]   = 1280
#define OFF_EH    (OVL + 41280)     // bf16 [160][64] = 20480
#define OFF_AGG   (OVL + 61760)     // bf16 [80][64]  = 10240 -> end 72000
// --- readout-phase overlay ---
#define OFF_MR    (OVL + 0)         // bf16 [8][128]  = 2048
#define OFF_H1    (OVL + 2048)      // bf16 [80][256] = 40960
#define OFF_H2    (OVL + 43008)     // bf16 [80][128] = 20480
#define OFF_H3    (OVL + 63488)     // bf16 [80][64]  = 10240
#define OFF_SC    (OVL + 73728)     // f32  [80]      = 320 -> end 74048
#define LDS_BYTES (OVL + 74048)     // 104768

// ---------------- workspace layout (bytes) ----------------
#define WS_WET 0        // bf16 64*160  = 20480
#define WS_WNT 20480    // bf16 64*96   = 12288
#define WS_W1T 32768    // bf16 256*192 = 98304
#define WS_W2T 131072   // bf16 128*256 = 65536
#define WS_W3T 196608   // bf16 64*128  = 16384

__device__ __forceinline__ v16bf cat16(v8bf lo, v8bf hi) {
  return __builtin_shufflevector(lo, hi, 0,1,2,3,4,5,6,7,8,9,10,11,12,13,14,15);
}
__device__ __forceinline__ v8bf zero8() {
  v8bf z;
#pragma unroll
  for (int i = 0; i < 8; ++i) z[i] = (__bf16)0.0f;
  return z;
}
__device__ __forceinline__ v8f zeroAcc() {
  v8f z;
#pragma unroll
  for (int i = 0; i < 8; ++i) z[i] = 0.0f;
  return z;
}
__device__ __forceinline__ v8f wmma_bf16(v16bf a, v16bf b, v8f c) {
  // D = A(16x32 bf16) * B(32x16 bf16) + C(16x16 f32)
  return __builtin_amdgcn_wmma_f32_16x16x32_bf16(
      /*neg_a=*/false, a, /*neg_b=*/false, b,
      /*c_mod=*/(short)0, c, /*reuse_a=*/false, /*reuse_b=*/false);
}
// B fragment: lane l -> column n0+(l&15); lanes 0-15 hold K=kk..kk+7,kk+16..kk+23,
// lanes 16-31 hold K=kk+8..kk+15,kk+24..kk+31 (K-contiguous weight storage, Kpad stride).
__device__ __forceinline__ v16bf ldB(const __bf16* wT, int Kpad, int n0,
                                     int l15, int hf, int kk) {
  const __bf16* p = wT + (n0 + l15) * Kpad + kk + hf * 8;
  return cat16(*(const v8bf*)p, *(const v8bf*)(p + 16));
}
// A-run selectors: 8 contiguous bf16 starting at concat-offset r for given row.
__device__ __forceinline__ v8bf g1run(const __bf16* nhb, const __bf16* ef,
                                      const __bf16* eh, const int* srcloc,
                                      int row, int r) {
  if (r < 64)  return *(const v8bf*)(nhb + srcloc[row] * HN + r);
  if (r < 72)  return *(const v8bf*)(ef + row * FE + (r - 64));
  if (r < 136) return *(const v8bf*)(eh + row * HE + (r - 72));
  return zero8();
}
__device__ __forceinline__ v8bf g2run(const __bf16* nf, const __bf16* agg,
                                      int row, int r) {
  if (r < 16) return *(const v8bf*)(nf + row * FN + r);
  if (r < 80) return *(const v8bf*)(agg + row * HN + (r - 16));
  return zero8();
}
__device__ __forceinline__ v8bf rorun(const __bf16* nhb, const __bf16* mr,
                                      int row, int r) {
  if (r < 64) return *(const v8bf*)(nhb + row * HN + r);
  return *(const v8bf*)(mr + (row / A_ATOMS) * MR + (r - 64));
}

// Weight transpose+convert: dst[n*Kpad + k] = (k<K) ? src[k*N + n] : 0 (bf16)
__global__ void transposeW(const float* __restrict__ src, __bf16* __restrict__ dst,
                           int K, int Kpad, int Ncols) {
  int idx = blockIdx.x * blockDim.x + threadIdx.x;
  if (idx >= Ncols * Kpad) return;
  int n = idx / Kpad, k = idx - n * Kpad;
  dst[idx] = (__bf16)((k < K) ? src[k * Ncols + n] : 0.0f);
}

__global__ void __launch_bounds__(THREADS) mpnn_kernel(
    const float* __restrict__ mol_reprs, const float* __restrict__ node_features,
    const float* __restrict__ edge_features, const int* __restrict__ edges,
    const __bf16* __restrict__ wET, const __bf16* __restrict__ wNT,
    const __bf16* __restrict__ w1T, const __bf16* __restrict__ w2T,
    const __bf16* __restrict__ w3T,
    const float* __restrict__ b_e, const float* __restrict__ b_n,
    const float* __restrict__ b1, const float* __restrict__ b2,
    const float* __restrict__ b3, const float* __restrict__ w4,
    const float* __restrict__ b4, float* __restrict__ out,
    int nTot, int maxAtoms) {
  extern __shared__ __align__(16) char lds[];
  __bf16* nhb  = (__bf16*)(lds + OFF_NHB);
  float*  nhf  = (float*)(lds + OFF_NHF);
  __bf16* wet  = (__bf16*)(lds + OFF_WET);
  __bf16* wnt  = (__bf16*)(lds + OFF_WNT);
  float*  beL  = (float*)(lds + OFF_BE);
  float*  bnL  = (float*)(lds + OFF_BN);
  __bf16* nf   = (__bf16*)(lds + OFF_NF);
  __bf16* ef   = (__bf16*)(lds + OFF_EF);
  int*  srcloc = (int*)(lds + OFF_SRC);
  int*  dstloc = (int*)(lds + OFF_DST);
  int*  cntL   = (int*)(lds + OFF_CNT);
  int*  inl    = (int*)(lds + OFF_INL);
  __bf16* eh   = (__bf16*)(lds + OFF_EH);
  __bf16* agg  = (__bf16*)(lds + OFF_AGG);

  const int tid  = threadIdx.x;
  const int lane = tid & 31;
  const int wv   = tid >> 5;     // wave id 0..7
  const int l15  = lane & 15;
  const int hf   = lane >> 4;    // lane half
  const int wg   = blockIdx.x;
  const int nodeBase = wg * NODES;   // 80 nodes per WG
  const int edgeBase = wg * NODES;   // 80 forward edges per WG

  // ---- stage inputs / weights into LDS ----
  {
    const uint32_t* s1 = (const uint32_t*)wET;  uint32_t* d1 = (uint32_t*)wet;
    for (int i = tid; i < (64 * K1PAD * 2) / 4; i += THREADS) d1[i] = s1[i];
    const uint32_t* s2 = (const uint32_t*)wNT;  uint32_t* d2 = (uint32_t*)wnt;
    for (int i = tid; i < (64 * K2PAD * 2) / 4; i += THREADS) d2[i] = s2[i];
  }
  for (int i = tid; i < HE; i += THREADS) beL[i] = b_e[i];
  for (int i = tid; i < HN; i += THREADS) bnL[i] = b_n[i];
  for (int i = tid; i < NODES * FN; i += THREADS)
    nf[i] = (__bf16)node_features[(size_t)nodeBase * FN + i];
  for (int i = tid; i < NODES * FE; i += THREADS) {
    ef[i] = (__bf16)edge_features[(size_t)edgeBase * FE + i];
    ef[NODES * FE + i] =
        (__bf16)edge_features[((size_t)nTot + edgeBase) * FE + i];
  }
  for (int e = tid; e < EDGES; e += THREADS) {
    int ge = (e < NODES) ? (edgeBase + e) : (nTot + edgeBase + (e - NODES));
    srcloc[e] = edges[2 * ge + 0] - nodeBase;
    dstloc[e] = edges[2 * ge + 1] - nodeBase;
  }
  for (int i = tid; i < NODES * HN; i += THREADS) nhb[i] = (__bf16)0.0f;
  for (int i = tid; i < EDGES * HE; i += THREADS) eh[i] = (__bf16)0.0f;
  for (int i = tid; i < NODES; i += THREADS) cntL[i] = 0;
  __syncthreads();
  for (int e = tid; e < EDGES; e += THREADS) {
    int d = dstloc[e];
    int slot = atomicAdd(&cntL[d], 1);
    if (slot < 4) inl[d * 4 + slot] = e;
  }
  __syncthreads();

  // ---- 8 message-passing iterations ----
  for (int it = 0; it < 8; ++it) {
    // GEMM1: [160 edges x 136(->160)] @ w_e -> edge_hidden (reads eh, so
    // hold accumulators across all waves, barrier, then store)
    v8f acc[5];
    int pm[5], pn[5];
#pragma unroll
    for (int pi = 0; pi < 5; ++pi) {
      int p = wv + pi * 8;            // 0..39: 10 M-tiles x 4 N-tiles
      pm[pi] = (p >> 2) * 16;
      pn[pi] = (p & 3) * 16;
      acc[pi] = zeroAcc();
    }
#pragma unroll
    for (int ks = 0; ks < 5; ++ks) {
      int kk = ks * 32;
#pragma unroll
      for (int pi = 0; pi < 5; ++pi) {
        int row = pm[pi] + l15;
        v8bf a0 = g1run(nhb, ef, eh, srcloc, row, kk + hf * 8);
        v8bf a1 = g1run(nhb, ef, eh, srcloc, row, kk + 16 + hf * 8);
        v16bf bF = ldB(wet, K1PAD, pn[pi], l15, hf, kk);
        acc[pi] = wmma_bf16(cat16(a0, a1), bF, acc[pi]);
      }
    }
    __syncthreads();   // everyone done READING eh/nhb
#pragma unroll
    for (int pi = 0; pi < 5; ++pi) {
      int col = pn[pi] + l15;
      float bb = beL[col];
#pragma unroll
      for (int v = 0; v < 8; ++v) {
        int row = pm[pi] + v + hf * 8;
        float x = acc[pi][v] + bb;
        eh[row * HE + col] = (__bf16)(x > 0.f ? x : 0.f);
      }
    }
    __syncthreads();
    // segment-sum of edge_hidden into agg (each node: <=4 in-edges)
    for (int idx = tid; idx < NODES * HN; idx += THREADS) {
      int n = idx >> 6, c = idx & 63;
      int k = cntL[n]; if (k > 4) k = 4;
      float s = 0.f;
      for (int j = 0; j < k; ++j) s += (float)eh[inl[n * 4 + j] * HE + c];
      agg[idx] = (__bf16)s;
    }
    __syncthreads();
    // GEMM2: [80 nodes x 80(->96)] @ w_n -> node_hidden (no read/write overlap)
    for (int p = wv; p < 20; p += 8) {
      int m0 = (p >> 2) * 16, n0 = (p & 3) * 16;
      v8f c2 = zeroAcc();
#pragma unroll
      for (int ks = 0; ks < 3; ++ks) {
        int kk = ks * 32;
        int row = m0 + l15;
        v8bf a0 = g2run(nf, agg, row, kk + hf * 8);
        v8bf a1 = g2run(nf, agg, row, kk + 16 + hf * 8);
        v16bf bF = ldB(wnt, K2PAD, n0, l15, hf, kk);
        c2 = wmma_bf16(cat16(a0, a1), bF, c2);
      }
      int col = n0 + l15;
      float bb = bnL[col];
#pragma unroll
      for (int v = 0; v < 8; ++v) {
        int row = m0 + v + hf * 8;
        float x = c2[v] + bb;
        x = x > 0.f ? x : 0.f;
        nhb[row * HN + col] = (__bf16)x;
        nhf[row * HN + col] = x;
      }
    }
    __syncthreads();
  }

  // ---- readout MLP (overlay reuses MP scratch; nhb/nhf persist) ----
  __bf16* mrL = (__bf16*)(lds + OFF_MR);
  __bf16* h1  = (__bf16*)(lds + OFF_H1);
  __bf16* h2  = (__bf16*)(lds + OFF_H2);
  __bf16* h3  = (__bf16*)(lds + OFF_H3);
  float* scoreL = (float*)(lds + OFF_SC);
  for (int i = tid; i < G_MOL * MR; i += THREADS)
    mrL[i] = (__bf16)mol_reprs[(size_t)(wg * G_MOL) * MR + i];
  __syncthreads();
  // layer1: [80 x 192] @ w1 -> 256
  for (int p = wv; p < 80; p += 8) {
    int m0 = (p >> 4) * 16, n0 = (p & 15) * 16;
    v8f c = zeroAcc();
#pragma unroll
    for (int ks = 0; ks < 6; ++ks) {
      int kk = ks * 32;
      int row = m0 + l15;
      v8bf a0 = rorun(nhb, mrL, row, kk + hf * 8);
      v8bf a1 = rorun(nhb, mrL, row, kk + 16 + hf * 8);
      v16bf bF = ldB(w1T, 192, n0, l15, hf, kk);
      c = wmma_bf16(cat16(a0, a1), bF, c);
    }
    int col = n0 + l15;
    float bb = b1[col];
#pragma unroll
    for (int v = 0; v < 8; ++v) {
      int row = m0 + v + hf * 8;
      float x = c[v] + bb;
      h1[row * L1 + col] = (__bf16)(x > 0.f ? x : 0.f);
    }
  }
  __syncthreads();
  // layer2: [80 x 256] -> 128
  for (int p = wv; p < 40; p += 8) {
    int m0 = (p >> 3) * 16, n0 = (p & 7) * 16;
    v8f c = zeroAcc();
#pragma unroll
    for (int ks = 0; ks < 8; ++ks) {
      int kk = ks * 32;
      int row = m0 + l15;
      const __bf16* base = h1 + row * L1;
      v8bf a0 = *(const v8bf*)(base + kk + hf * 8);
      v8bf a1 = *(const v8bf*)(base + kk + 16 + hf * 8);
      v16bf bF = ldB(w2T, 256, n0, l15, hf, kk);
      c = wmma_bf16(cat16(a0, a1), bF, c);
    }
    int col = n0 + l15;
    float bb = b2[col];
#pragma unroll
    for (int v = 0; v < 8; ++v) {
      int row = m0 + v + hf * 8;
      float x = c[v] + bb;
      h2[row * L2 + col] = (__bf16)(x > 0.f ? x : 0.f);
    }
  }
  __syncthreads();
  // layer3: [80 x 128] -> 64
  for (int p = wv; p < 20; p += 8) {
    int m0 = (p >> 2) * 16, n0 = (p & 3) * 16;
    v8f c = zeroAcc();
#pragma unroll
    for (int ks = 0; ks < 4; ++ks) {
      int kk = ks * 32;
      int row = m0 + l15;
      const __bf16* base = h2 + row * L2;
      v8bf a0 = *(const v8bf*)(base + kk + hf * 8);
      v8bf a1 = *(const v8bf*)(base + kk + 16 + hf * 8);
      v16bf bF = ldB(w3T, 128, n0, l15, hf, kk);
      c = wmma_bf16(cat16(a0, a1), bF, c);
    }
    int col = n0 + l15;
    float bb = b3[col];
#pragma unroll
    for (int v = 0; v < 8; ++v) {
      int row = m0 + v + hf * 8;
      float x = c[v] + bb;
      h3[row * L3 + col] = (__bf16)(x > 0.f ? x : 0.f);
    }
  }
  __syncthreads();
  // layer4 dot + sigmoid
  for (int n = tid; n < NODES; n += THREADS) {
    float dot = b4[0];
    for (int k = 0; k < L3; ++k) dot += (float)h3[n * L3 + k] * w4[k];
    scoreL[n] = 1.0f / (1.0f + __expf(-dot));
  }
  __syncthreads();
  // gated scatter to (B, maxAtoms, 64); pad rows written as zero
  const int perMol = maxAtoms * HN;
  const int outPerWG = G_MOL * perMol;
  const size_t outBase = (size_t)wg * outPerWG;
  for (int idx = tid; idx < outPerWG; idx += THREADS) {
    int molL = idx / perMol;
    int rem = idx - molL * perMol;
    int pos = rem / HN;
    int c = rem & (HN - 1);
    float v = 0.0f;
    if (pos < A_ATOMS) {
      int nloc = molL * A_ATOMS + pos;
      v = nhf[nloc * HN + c] * scoreL[nloc];
    }
    out[outBase + idx] = v;
  }
}

extern "C" void kernel_launch(void* const* d_in, const int* in_sizes, int n_in,
                              void* d_out, int out_size, void* d_ws, size_t ws_size,
                              hipStream_t stream) {
  const float* mol_reprs     = (const float*)d_in[0];
  const float* node_features = (const float*)d_in[1];
  const float* edge_features = (const float*)d_in[2];
  const int*   edges         = (const int*)d_in[5];
  const float* w_e = (const float*)d_in[8];
  const float* b_e = (const float*)d_in[9];
  const float* w_n = (const float*)d_in[10];
  const float* b_n = (const float*)d_in[11];
  const float* w1  = (const float*)d_in[12];
  const float* b1  = (const float*)d_in[13];
  const float* w2  = (const float*)d_in[14];
  const float* b2  = (const float*)d_in[15];
  const float* w3  = (const float*)d_in[16];
  const float* b3  = (const float*)d_in[17];
  const float* w4  = (const float*)d_in[18];
  const float* b4  = (const float*)d_in[19];
  float* out = (float*)d_out;

  const int numMol   = in_sizes[0] / MR;         // 16384
  const int nTot     = in_sizes[1] / FN;         // total nodes
  const int maxAtoms = out_size / (numMol * HN); // 12

  char* ws = (char*)d_ws;
  __bf16* wET = (__bf16*)(ws + WS_WET);
  __bf16* wNT = (__bf16*)(ws + WS_WNT);
  __bf16* w1T = (__bf16*)(ws + WS_W1T);
  __bf16* w2T = (__bf16*)(ws + WS_W2T);
  __bf16* w3T = (__bf16*)(ws + WS_W3T);

  auto launchT = [&](const float* s, __bf16* d, int K, int Kpad, int Nc) {
    int tot = Nc * Kpad;
    transposeW<<<(tot + 255) / 256, 256, 0, stream>>>(s, d, K, Kpad, Nc);
  };
  launchT(w_e, wET, 136, K1PAD, 64);
  launchT(w_n, wNT, 80, K2PAD, 64);
  launchT(w1, w1T, 192, 192, 256);
  launchT(w2, w2T, 256, 256, 128);
  launchT(w3, w3T, 128, 128, 64);

  const int nWG = numMol / G_MOL;
  mpnn_kernel<<<nWG, THREADS, LDS_BYTES, stream>>>(
      mol_reprs, node_features, edge_features, edges,
      wET, wNT, w1T, w2T, w3T,
      b_e, b_n, b1, b2, b3, w4, b4, out, nTot, maxAtoms);
}
